// Temporal_Attention_28656021799218
// MI455X (gfx1250) — compile-verified, matrix operand order checked by reading, not run on hardware
//
#include <hip/hip_runtime.h>

// ---------------------------------------------------------------------------
// Fused ConvLSTM-style recurrence for MI455X (gfx1250), wave32 + WMMA bf16.
//
// Shapes: B=8, T-1=16, C=64, S=32, H=128, N = B*S*S = 8192, 4H = 512.
// One workgroup = 16 rows of N, all 16 timesteps (recurrence kept on-chip).
// Roofline: 25.8 GFLOP of GEMM, ~100 MB mandatory HBM traffic (4.3 us at
// 23.3 TB/s) -> must use WMMA (bf16 in, f32 accum) and never materialize the
// 268 MB x_gates tensor in HBM. Weights are WGP-resident in LDS as bf16.
//
// Pipeline per step t (2 barriers):
//   store xr(t) regs -> LDS bf16 ; issue global loads xr(t+1) ; barrier
//   24x v_wmma_f32_16x16x32_bf16 per wave (K=64 x-part + K=128 h-part)
//   spill gates (transposed, ds_store_b128) ; barrier
//   elementwise sigmoid/tanh, c-state in regs, h -> LDS bf16 + global f32
// ---------------------------------------------------------------------------

typedef __attribute__((ext_vector_type(16))) __bf16 v16bf;
typedef __attribute__((ext_vector_type(8)))  float  v8f;
typedef __attribute__((ext_vector_type(4)))  float  v4f;

#define B_      8
#define T_      16
#define C_      64
#define S2_     1024      // S*S
#define H_      128
#define G4_     512       // 4*H
#define NROWS   16        // rows of N per workgroup
#define NTHR    256       // 8 waves

// LDS strides (elements), padded for bank-conflict avoidance; bf16 strides are
// multiples of 16 elements (32B) so v16bf fragment loads stay aligned.
#define WIH_S   80        // W_ih row stride  (64 used)
#define WHH_S   144       // W_hh row stride  (128 used)
#define H_S     144       // h row stride     (128 used)
#define X_S     80        // x row stride     (64 used)
#define GT_S    20        // TRANSPOSED gate staging: [col g][row m], 16 used

#define SMEM_BYTES ((size_t)( (G4_*WIH_S + G4_*WHH_S + NROWS*H_S + NROWS*X_S) * 2 \
                            +  G4_ * GT_S * 4 ))

__device__ __forceinline__ unsigned short f2bf(float f) {
  unsigned int u = __float_as_uint(f);
  u += 0x7FFFu + ((u >> 16) & 1u);            // round-to-nearest-even
  return (unsigned short)(u >> 16);
}
__device__ __forceinline__ float sigmoid_f(float x) {
  return 1.0f / (1.0f + __expf(-x));
}
__device__ __forceinline__ float tanh_f(float x) {
  float e = __expf(2.0f * x);
  return (e - 1.0f) / (e + 1.0f);
}

__global__ __launch_bounds__(NTHR)
void fused_lstm_wmma(const float* __restrict__ x,     // [B,T,C,S,S]
                     const float* __restrict__ Wih,   // [4H,C]
                     const float* __restrict__ Whh,   // [4H,H]
                     const float* __restrict__ bih,   // [4H]
                     const float* __restrict__ bhh,   // [4H]
                     float* __restrict__ out) {       // [B,T,H,S,S]
  extern __shared__ char smem_raw[];
  unsigned short* lds_wih = (unsigned short*)smem_raw;            // G4_*WIH_S
  unsigned short* lds_whh = lds_wih + G4_ * WIH_S;                // G4_*WHH_S
  unsigned short* lds_h   = lds_whh + G4_ * WHH_S;                // NROWS*H_S
  unsigned short* lds_x   = lds_h   + NROWS * H_S;                // NROWS*X_S
  float*          lds_g   = (float*)(lds_x + NROWS * X_S);        // G4_*GT_S

  const int tid  = threadIdx.x;
  const int lane = tid & 31;
  const int wave = tid >> 5;                 // 0..7 -> gate cols [wave*64, +64)
  const int nsel  = lane & 15;               // N (col) within a 16x16 tile
  const int khalf = (lane >> 4) << 4;        // 0 or 16: K half, contiguous k = 16h + s (A and B alike)
  const int r0    = (lane >> 4) << 3;        // C/D rows r0..r0+7 in this lane

  // ---- one-time: weights -> LDS bf16, h := 0 -------------------------------
  for (int i = tid; i < G4_ * C_; i += NTHR) {
    int g = i >> 6, c = i & 63;
    lds_wih[g * WIH_S + c] = f2bf(Wih[i]);
  }
  for (int i = tid; i < G4_ * H_; i += NTHR) {
    int g = i >> 7, c = i & 127;
    lds_whh[g * WHH_S + c] = f2bf(Whh[i]);
  }
  for (int i = tid; i < NROWS * H_S; i += NTHR) lds_h[i] = (unsigned short)0;

  // fused bias per owned gate column (constant over rows & steps)
  float bias[4];
#pragma unroll
  for (int ct = 0; ct < 4; ++ct) {
    int g = wave * 64 + ct * 16 + nsel;
    bias[ct] = bih[g] + bhh[g];
  }

  // ---- per-thread x-tile plan (4 elements/thread/step) ---------------------
  const int rowStart = blockIdx.x * NROWS;   // base row in N
  long xbase[4]; int xdst[4];
#pragma unroll
  for (int k = 0; k < 4; ++k) {
    int idx = tid + k * NTHR;                // 0..1023
    int row = idx >> 6, c = idx & 63;
    int n = rowStart + row;
    int b = n >> 10, sp = n & 1023;
    xbase[k] = ((long)(b * T_) * C_ + c) * S2_ + sp;  // + t*C_*S2_ per step
    xdst[k]  = row * X_S + c;
  }

  // ---- elementwise-phase ownership: row = tid&15, 8 hidden cols ------------
  const int erow   = tid & 15;
  const int ecol0  = (tid >> 4) << 3;        // 0,8,...,120
  const int en     = rowStart + erow;
  const int eb     = en >> 10, esp = en & 1023;
  float cst[8];
#pragma unroll
  for (int e = 0; e < 8; ++e) cst[e] = 0.0f;

  // software pipeline: preload x(0) into registers
  float xr[4];
#pragma unroll
  for (int k = 0; k < 4; ++k) xr[k] = x[xbase[k]];

  __syncthreads();                           // weights + h ready

  for (int t = 0; t < T_; ++t) {
    // -- stage x(t) regs -> LDS bf16; issue x(t+1) loads (overlap full step) -
#pragma unroll
    for (int k = 0; k < 4; ++k) lds_x[xdst[k]] = f2bf(xr[k]);
    if (t + 1 < T_) {
#pragma unroll
      for (int k = 0; k < 4; ++k)
        xr[k] = x[xbase[k] + (long)(t + 1) * (C_ * S2_)];
    }
    if (t + 2 < T_) {                        // pre-stage L2 (global_prefetch)
      __builtin_prefetch(&x[xbase[0] + (long)(t + 2) * (C_ * S2_)], 0, 1);
    }
    __syncthreads();

    // -- WMMA: gates(16x512) = x(16x64)@Wih^T + h(16x128)@Whh^T + bias -------
    v8f acc[4];
#pragma unroll
    for (int ct = 0; ct < 4; ++ct) {
      v8f a;
#pragma unroll
      for (int r = 0; r < 8; ++r) a[r] = bias[ct];
      acc[ct] = a;
    }
#pragma unroll
    for (int kt = 0; kt < 2; ++kt) {         // K = 64 over x
      v16bf afrag = *(const v16bf*)&lds_x[nsel * X_S + kt * 32 + khalf];
#pragma unroll
      for (int ct = 0; ct < 4; ++ct) {
        int g = wave * 64 + ct * 16 + nsel;
        v16bf bfrag = *(const v16bf*)&lds_wih[g * WIH_S + kt * 32 + khalf];
        acc[ct] = __builtin_amdgcn_wmma_f32_16x16x32_bf16(
            false, afrag, false, bfrag, (short)0, acc[ct], false, false);
      }
    }
#pragma unroll
    for (int kt = 0; kt < 4; ++kt) {         // K = 128 over h
      v16bf afrag = *(const v16bf*)&lds_h[nsel * H_S + kt * 32 + khalf];
#pragma unroll
      for (int ct = 0; ct < 4; ++ct) {
        int g = wave * 64 + ct * 16 + nsel;
        v16bf bfrag = *(const v16bf*)&lds_whh[g * WHH_S + kt * 32 + khalf];
        acc[ct] = __builtin_amdgcn_wmma_f32_16x16x32_bf16(
            false, afrag, false, bfrag, (short)0, acc[ct], false, false);
      }
    }
    // -- spill gates TRANSPOSED [g][m]: per lane 8 contiguous rows of one
    //    column -> two ds_store_b128 per col-tile (8 per wave total) ---------
#pragma unroll
    for (int ct = 0; ct < 4; ++ct) {
      int g = wave * 64 + ct * 16 + nsel;
      v4f lo, hi;
#pragma unroll
      for (int r = 0; r < 4; ++r) { lo[r] = acc[ct][r]; hi[r] = acc[ct][r + 4]; }
      *(v4f*)&lds_g[g * GT_S + r0]     = lo;
      *(v4f*)&lds_g[g * GT_S + r0 + 4] = hi;
    }
    __syncthreads();

    // -- elementwise LSTM cell update ---------------------------------------
    long obase = ((long)(eb * T_ + t) * H_) * S2_ + esp;
#pragma unroll
    for (int e = 0; e < 8; ++e) {
      int hc = ecol0 + e;
      float ig = sigmoid_f(lds_g[(hc)          * GT_S + erow]);
      float fg = sigmoid_f(lds_g[(H_ + hc)     * GT_S + erow]);
      float gg = tanh_f   (lds_g[(2 * H_ + hc) * GT_S + erow]);
      float og = sigmoid_f(lds_g[(3 * H_ + hc) * GT_S + erow]);
      float cn = fg * cst[e] + ig * gg;
      cst[e] = cn;
      float hn = og * tanh_f(cn);
      lds_h[erow * H_S + hc] = f2bf(hn);           // feed next step (bf16)
      out[obase + (long)hc * S2_] = hn;            // [b,t,h,s,s] fp32
    }
    // NOTE: no trailing barrier needed -- the top-of-loop barrier already orders
    // lds_h writes (here) against next iteration's WMMA reads (after the
    // next top-of-loop barrier), and lds_x stores vs. WMMA reads are ordered
    // by the gates barrier of this iteration.
  }
}

extern "C" void kernel_launch(void* const* d_in, const int* in_sizes, int n_in,
                              void* d_out, int out_size, void* d_ws, size_t ws_size,
                              hipStream_t stream) {
  const float* x   = (const float*)d_in[0];
  const float* Wih = (const float*)d_in[1];
  const float* Whh = (const float*)d_in[2];
  const float* bih = (const float*)d_in[3];
  const float* bhh = (const float*)d_in[4];
  float* out = (float*)d_out;

  const int n_blocks = (B_ * S2_) / NROWS;   // 8192/16 = 512 workgroups
  fused_lstm_wmma<<<dim3(n_blocks), dim3(NTHR), SMEM_BYTES, stream>>>(
      x, Wih, Whh, bih, bhh, out);
}
